// GPT_67619965108241
// MI455X (gfx1250) — matrix operand order repaired and checked by reading, not compiled, not run on hardware
//
#include <hip/hip_runtime.h>

// ---------------------------------------------------------------------------
// GPT-2 forward for MI455X (gfx1250), wave32, WMMA bf16 path.
// - All GEMMs + attention matmuls: V_WMMA_F32_16X16X32_BF16 (fp32 accum).
// - Straight-copy tiles (GEMM A, attention K) staged with
//   GLOBAL_LOAD_ASYNC_TO_LDS_B128 + s_wait_asynccnt (CDNA5 async path).
// - fp32 weights converted to bf16 during LDS staging (weights are
//   L2-resident at 192MB, so fp32 re-reads across M-tiles hit L2).
// - Flash attention (online softmax) avoids 2.4GB of score traffic.
// - Logits (823MB fp32) written with nontemporal stores to protect L2.
// ---------------------------------------------------------------------------

typedef __bf16 bf16_t;
typedef __attribute__((ext_vector_type(16))) __bf16 v16bf;
typedef __attribute__((ext_vector_type(8)))  __bf16 v8bf;
typedef __attribute__((ext_vector_type(8)))  float  v8f;

// Async 16B global->LDS copy (no VGPR data path, ASYNCcnt-tracked).
// LDS operand = low 32 bits of generic pointer (flat->LDS uses addr[31:0]).
__device__ __forceinline__ void async_ld_b128(void* ldst, const void* src) {
  asm volatile("global_load_async_to_lds_b128 %0, %1, off"
               :: "v"((unsigned)(size_t)ldst), "v"(src)
               : "memory");
}
__device__ __forceinline__ void wait_async0() {
  asm volatile("s_wait_asynccnt 0" ::: "memory");
}

// Build a 16-wide bf16 fragment from two 16B LDS chunks (two ds_load_b128).
__device__ __forceinline__ v16bf ld_frag(const bf16_t* p0, const bf16_t* p1) {
  v8bf lo = *(const v8bf*)p0;
  v8bf hi = *(const v8bf*)p1;
  return __builtin_shufflevector(lo, hi, 0,1,2,3,4,5,6,7,8,9,10,11,12,13,14,15);
}

__device__ __forceinline__ float half_max(float v) {   // reduce within 16-lane half
  v = fmaxf(v, __shfl_xor(v, 1, 32));
  v = fmaxf(v, __shfl_xor(v, 2, 32));
  v = fmaxf(v, __shfl_xor(v, 4, 32));
  v = fmaxf(v, __shfl_xor(v, 8, 32));
  return v;
}
__device__ __forceinline__ float half_sum(float v) {
  v += __shfl_xor(v, 1, 32);
  v += __shfl_xor(v, 2, 32);
  v += __shfl_xor(v, 4, 32);
  v += __shfl_xor(v, 8, 32);
  return v;
}

// ---------------------------------------------------------------------------
// Embedding: x[row,:] = wte[tok,:] + wpe[0,:]   (reference uses only wpe[0])
// ---------------------------------------------------------------------------
__global__ __launch_bounds__(256) void embed_k(const int* __restrict__ tokens,
                                               const float* __restrict__ wte,
                                               const float* __restrict__ wpe,
                                               float* __restrict__ x) {
  int row = blockIdx.x, t = threadIdx.x;
  int tok = tokens[row];
  const float* wr = wte + (size_t)tok * 768;
  float* xr = x + (size_t)row * 768;
  xr[t]       = wr[t]       + wpe[t];
  xr[t + 256] = wr[t + 256] + wpe[t + 256];
  xr[t + 512] = wr[t + 512] + wpe[t + 512];
}

// ---------------------------------------------------------------------------
// LayerNorm over E=768, fp32 in -> bf16 out. One block per row, 256 threads.
// ---------------------------------------------------------------------------
__global__ __launch_bounds__(256) void layernorm_k(const float* __restrict__ x,
                                                   const float* __restrict__ w,
                                                   const float* __restrict__ b,
                                                   bf16_t* __restrict__ out) {
  const int E = 768;
  int row = blockIdx.x, t = threadIdx.x;
  const float* xr = x + (size_t)row * E;
  __shared__ float red[256];
  float v0 = xr[t], v1 = xr[t + 256], v2 = xr[t + 512];
  red[t] = v0 + v1 + v2;
  __syncthreads();
  for (int s = 128; s > 0; s >>= 1) { if (t < s) red[t] += red[t + s]; __syncthreads(); }
  float mean = red[0] * (1.0f / 768.0f);
  __syncthreads();
  float d0 = v0 - mean, d1 = v1 - mean, d2 = v2 - mean;
  red[t] = d0 * d0 + d1 * d1 + d2 * d2;
  __syncthreads();
  for (int s = 128; s > 0; s >>= 1) { if (t < s) red[t] += red[t + s]; __syncthreads(); }
  float rstd = rsqrtf(red[0] * (1.0f / 768.0f) + 1e-5f);
  bf16_t* orow = out + (size_t)row * E;
  orow[t]       = (bf16_t)(d0 * rstd * w[t]       + b[t]);
  orow[t + 256] = (bf16_t)(d1 * rstd * w[t + 256] + b[t + 256]);
  orow[t + 512] = (bf16_t)(d2 * rstd * w[t + 512] + b[t + 512]);
}

// ---------------------------------------------------------------------------
// Tiled WMMA GEMM:  C[M,N] = A_bf16[M,K] * W_f32 (+bias) with fused epilogue.
//   WT_NMAJOR=false: W stored [K,N] (qkv/atp/fc/fp weights); N % 128 == 0.
//   WT_NMAJOR=true : W stored [N,K] (lm_w: torch Linear [out,in]).
// MODE 0: bf16 out (+bias)        MODE 1: gelu -> bf16 out (+bias)
// MODE 2: fp32 residual  x += acc+bias        MODE 3: fp32 out (logits, NT)
// Block: 128x128 tile, BK=32, 8 waves -> each wave 64x32 (4x2 WMMA frags).
// ---------------------------------------------------------------------------
template <int MODE, bool WT_NMAJOR>
__global__ __launch_bounds__(256) void gemm_wmma(const bf16_t* __restrict__ A,
                                                 const float* __restrict__ W,
                                                 const float* __restrict__ bias,
                                                 float* __restrict__ outF,
                                                 bf16_t* __restrict__ outB,
                                                 int M, int N, int K, int ldw) {
  const int SAS = 48;                     // padded LDS row stride (bf16 elems)
  __shared__ bf16_t smem[2 * 128 * SAS];
  bf16_t* sA = smem;                      // [m][k]
  bf16_t* sB = smem + 128 * SAS;          // [n][k] (transposed -> b128 frags)

  int t = threadIdx.x;
  int lane = t & 31, wv = t >> 5;
  int wm = wv & 1, wn = wv >> 1;          // wave: M half (0..1), N quarter (0..3)
  int lm = lane & 15, lh = lane >> 4;
  int rowBase = blockIdx.y * 128;
  int colBase = blockIdx.x * 128;

  v8f acc[4][2] = {};

  for (int kk = 0; kk < K; kk += 32) {
    __syncthreads();
    // ---- stage A tile: pure copy -> async global->LDS (2 x 16B/thread) ----
    {
      int ar = t >> 1, ac = (t & 1) * 16;
      const bf16_t* gp = A + (size_t)(rowBase + ar) * K + kk + ac;
      bf16_t* lp = sA + ar * SAS + ac;
      async_ld_b128(lp, gp);
      async_ld_b128(lp + 8, gp + 8);
    }
    // ---- stage W tile, convert fp32->bf16, transposed [n][k] ----
    if (WT_NMAJOR) {
      // W[N][K]: contiguous along K; per-thread 16 contiguous floats.
      int n = t >> 1, k0 = (t & 1) * 16;
      int col = colBase + n;
      alignas(16) bf16_t tmp[16];
      if (col < N) {
#pragma unroll
        for (int e = 0; e < 16; ++e)
          tmp[e] = (bf16_t)W[(size_t)col * ldw + (kk + k0 + e)];
      } else {
#pragma unroll
        for (int e = 0; e < 16; ++e) tmp[e] = (bf16_t)0.0f;
      }
      *(v8bf*)(sB + n * SAS + k0)     = *(const v8bf*)tmp;
      *(v8bf*)(sB + n * SAS + k0 + 8) = *(const v8bf*)(tmp + 8);
    } else {
      // W[K][N]: coalesced float4 along N, 4-way bf16 transpose scatter.
      // (all [K,N] GEMMs here have N % 128 == 0 -> no column guard needed)
#pragma unroll
      for (int r = 0; r < 4; ++r) {
        int idx = r * 256 + t;            // 1024 float4 groups
        int k = idx >> 5;                 // 0..31
        int n4 = (idx & 31) * 4;          // 0..124
        float4 wvv = *(const float4*)(W + (size_t)(kk + k) * ldw + colBase + n4);
        sB[(n4 + 0) * SAS + k] = (bf16_t)wvv.x;
        sB[(n4 + 1) * SAS + k] = (bf16_t)wvv.y;
        sB[(n4 + 2) * SAS + k] = (bf16_t)wvv.z;
        sB[(n4 + 3) * SAS + k] = (bf16_t)wvv.w;
      }
    }
    wait_async0();
    __syncthreads();

    // ---- fragments + 8 WMMAs ----
    v16bf af[4], bfr[2];
#pragma unroll
    for (int mi = 0; mi < 4; ++mi) {
      const bf16_t* base = sA + (wm * 64 + mi * 16 + lm) * SAS;
      af[mi] = ld_frag(base + lh * 8, base + 16 + lh * 8);
    }
#pragma unroll
    for (int ni = 0; ni < 2; ++ni) {
      const bf16_t* base = sB + (wn * 32 + ni * 16 + lm) * SAS;
      bfr[ni] = ld_frag(base + lh * 8, base + 16 + lh * 8);
    }
#pragma unroll
    for (int mi = 0; mi < 4; ++mi)
#pragma unroll
      for (int ni = 0; ni < 2; ++ni)
        acc[mi][ni] = __builtin_amdgcn_wmma_f32_16x16x32_bf16(
            false, af[mi], false, bfr[ni], (short)0, acc[mi][ni], false, false);
  }

  // ---- epilogue: C layout lane-> N=lm, VGPR v -> M=lh*8+v ----
#pragma unroll
  for (int mi = 0; mi < 4; ++mi) {
#pragma unroll
    for (int ni = 0; ni < 2; ++ni) {
      int col = colBase + wn * 32 + ni * 16 + lm;
      if (col >= N) continue;
      float bv = bias ? bias[col] : 0.0f;
      v8f r = acc[mi][ni];
#pragma unroll
      for (int v = 0; v < 8; ++v) {
        int row = rowBase + wm * 64 + mi * 16 + lh * 8 + v;
        float val = r[v] + bv;
        size_t idx = (size_t)row * (size_t)N + col;
        if (MODE == 0) {
          outB[idx] = (bf16_t)val;
        } else if (MODE == 1) {
          float g = 0.5f * val *
                    (1.0f + tanhf(0.7978845608028654f * (val + 0.044715f * val * val * val)));
          outB[idx] = (bf16_t)g;
        } else if (MODE == 2) {
          outF[idx] += val;                         // residual add into x (fp32)
        } else {
          __builtin_nontemporal_store(val, &outF[idx]);  // logits: don't pollute L2
        }
      }
    }
  }
}

// ---------------------------------------------------------------------------
// Flash attention. qkv bf16 [B*T, 3E] (q|k|v, head h at col h*64).
// Block = (128 queries, head, batch); 8 waves, one 16-query tile each.
// Streams 32-key tiles; S=Q*K^T and O+=P*V via WMMA; online softmax with
// shfl_xor row reductions; P reshaped C-layout->A-layout through LDS.
// K tile staged via async global->LDS; V tile needs a transpose (VGPR path).
// ---------------------------------------------------------------------------
__global__ __launch_bounds__(256) void attn_flash(const bf16_t* __restrict__ qkv,
                                                  bf16_t* __restrict__ y) {
  const int T = 1024, C3 = 2304, E = 768;
  int qb = blockIdx.x, h = blockIdx.y, b = blockIdx.z;
  int t = threadIdx.x, lane = t & 31, w = t >> 5;
  int lm = lane & 15, lh = lane >> 4;

  __shared__ bf16_t smem[32 * 72 + 64 * 40 + 8 * 16 * 40];
  bf16_t* sK = smem;                    // [key][d]  -> S B-frags (b128)
  bf16_t* sV = smem + 32 * 72;          // [d][key]  -> PV B-frags (b128)
  bf16_t* sP = sV + 64 * 40;            // per-wave P bounce [m][key]
  bf16_t* sPw = sP + w * 16 * 40;

  int qt = qb * 128 + w * 16;           // this wave's first query token
  size_t rowBase = (size_t)b * T;
  int qOff = h * 64, kOff = 768 + h * 64, vOff = 1536 + h * 64;

  // Q fragments (A layout), loaded once: two 32-wide K-chunks of D=64.
  v16bf qa[2];
  {
    const bf16_t* qp = qkv + (rowBase + qt + lm) * C3 + qOff;
#pragma unroll
    for (int c = 0; c < 2; ++c)
      qa[c] = ld_frag(qp + c * 32 + lh * 8, qp + c * 32 + 16 + lh * 8);
  }

  v8f o[4] = {};
  float mstat[8], lstat[8], corr[8];
#pragma unroll
  for (int v = 0; v < 8; ++v) { mstat[v] = -1e30f; lstat[v] = 0.0f; }

  const float scale = 0.03608439182435161f;  // 1/sqrt(768) per reference
  int ktMax = (qb + 1) * 4;                  // causal upper bound for block

  for (int kt = 0; kt < ktMax; ++kt) {
    __syncthreads();
    // ---- cooperative staging of 32-key K (async) and V (transpose) ----
    {
      int key = t >> 3, d0 = (t & 7) * 8;
      size_t krow = (rowBase + kt * 32 + key) * C3;
      async_ld_b128(sK + key * 72 + d0, qkv + krow + kOff + d0);
      v8bf vvec = *(const v8bf*)(qkv + krow + vOff + d0);
#pragma unroll
      for (int e = 0; e < 8; ++e) sV[(d0 + e) * 40 + key] = vvec[e];
    }
    wait_async0();
    __syncthreads();
    if (kt * 32 > qt + 15) continue;  // inactive wave: staging + barriers only

    // ---- S = Q*K^T for two 16-key sub-tiles ----
    float s[2][8];
#pragma unroll
    for (int f = 0; f < 2; ++f) {
      v8f sacc = {};
#pragma unroll
      for (int c = 0; c < 2; ++c) {
        const bf16_t* kp = sK + (f * 16 + lm) * 72 + c * 32;
        v16bf kb = ld_frag(kp + lh * 8, kp + 16 + lh * 8);
        sacc = __builtin_amdgcn_wmma_f32_16x16x32_bf16(
            false, qa[c], false, kb, (short)0, sacc, false, false);
      }
      int kg = kt * 32 + f * 16 + lm;
#pragma unroll
      for (int v = 0; v < 8; ++v) {
        int qg = qt + lh * 8 + v;
        s[f][v] = (kg <= qg) ? sacc[v] * scale : -1e30f;
      }
    }
    // ---- online softmax update (row = lane-half x VGPR slot) ----
#pragma unroll
    for (int v = 0; v < 8; ++v) {
      float mt = half_max(fmaxf(s[0][v], s[1][v]));
      float mn = fmaxf(mstat[v], mt);
      corr[v] = __expf(mstat[v] - mn);
      float p0 = __expf(s[0][v] - mn);
      float p1 = __expf(s[1][v] - mn);
      s[0][v] = p0; s[1][v] = p1;
      lstat[v] = lstat[v] * corr[v] + half_sum(p0 + p1);
      mstat[v] = mn;
    }
    // ---- P: C-layout -> A-layout via private LDS bounce ----
#pragma unroll
    for (int f = 0; f < 2; ++f)
#pragma unroll
      for (int v = 0; v < 8; ++v)
        sPw[(lh * 8 + v) * 40 + f * 16 + lm] = (bf16_t)s[f][v];
    v16bf pf;
    {
      const bf16_t* pp = sPw + lm * 40;
      pf = ld_frag(pp + lh * 8, pp + 16 + lh * 8);
    }
    // ---- O = O*corr + P*V over 4 d-chunks of 16 ----
#pragma unroll
    for (int dn = 0; dn < 4; ++dn) {
#pragma unroll
      for (int v = 0; v < 8; ++v) o[dn][v] *= corr[v];
      const bf16_t* vp = sV + (dn * 16 + lm) * 40;
      v16bf vb = ld_frag(vp + lh * 8, vp + 16 + lh * 8);
      o[dn] = __builtin_amdgcn_wmma_f32_16x16x32_bf16(
          false, pf, false, vb, (short)0, o[dn], false, false);
    }
  }

  // ---- normalize and write y (bf16) ----
#pragma unroll
  for (int v = 0; v < 8; ++v) lstat[v] = 1.0f / lstat[v];
#pragma unroll
  for (int dn = 0; dn < 4; ++dn)
#pragma unroll
    for (int v = 0; v < 8; ++v) {
      int tok = qt + lh * 8 + v;
      y[(rowBase + tok) * E + h * 64 + dn * 16 + lm] = (bf16_t)(o[dn][v] * lstat[v]);
    }
}

// ---------------------------------------------------------------------------
// Host-side orchestration.
// ---------------------------------------------------------------------------
extern "C" void kernel_launch(void* const* d_in, const int* in_sizes, int n_in,
                              void* d_out, int out_size, void* d_ws, size_t ws_size,
                              hipStream_t stream) {
  const int B = 4, T = 1024, L = 12, E = 768, FF = 3072, C3 = 2304, V = 50257;
  const int M = B * T;  // 4096

  const int*   tokens = (const int*)d_in[0];
  const float* wte    = (const float*)d_in[1];
  const float* wpe    = (const float*)d_in[2];
  const float* ln1_w  = (const float*)d_in[3];
  const float* ln1_b  = (const float*)d_in[4];
  const float* qkv_w  = (const float*)d_in[5];
  const float* qkv_b  = (const float*)d_in[6];
  const float* atp_w  = (const float*)d_in[7];
  const float* atp_b  = (const float*)d_in[8];
  const float* ln2_w  = (const float*)d_in[9];
  const float* ln2_b  = (const float*)d_in[10];
  const float* fc_w   = (const float*)d_in[11];
  const float* fc_b   = (const float*)d_in[12];
  const float* fp_w   = (const float*)d_in[13];
  const float* fp_b   = (const float*)d_in[14];
  const float* lnf_w  = (const float*)d_in[15];
  const float* lnf_b  = (const float*)d_in[16];
  const float* lm_w   = (const float*)d_in[17];

  // Workspace carve (~69 MB total).
  char* p = (char*)d_ws;
  float*  x    = (float*)p;  p += (size_t)M * E * sizeof(float);    // residual stream
  bf16_t* hb   = (bf16_t*)p; p += (size_t)M * E * sizeof(bf16_t);   // LN output
  bf16_t* qkvb = (bf16_t*)p; p += (size_t)M * C3 * sizeof(bf16_t);  // q|k|v
  bf16_t* yb   = (bf16_t*)p; p += (size_t)M * E * sizeof(bf16_t);   // attn output
  bf16_t* mb   = (bf16_t*)p; p += (size_t)M * FF * sizeof(bf16_t);  // MLP hidden
  (void)ws_size; (void)n_in; (void)in_sizes; (void)out_size;

  dim3 blk(256);
  embed_k<<<M, blk, 0, stream>>>(tokens, wte, wpe, x);

  for (int i = 0; i < L; ++i) {
    layernorm_k<<<M, blk, 0, stream>>>(x, ln1_w + (size_t)i * E, ln1_b + (size_t)i * E, hb);

    gemm_wmma<0, false><<<dim3(C3 / 128, M / 128), blk, 0, stream>>>(
        hb, qkv_w + (size_t)i * E * C3, qkv_b + (size_t)i * C3,
        nullptr, qkvb, M, C3, E, C3);

    attn_flash<<<dim3(T / 128, 12, B), blk, 0, stream>>>(qkvb, yb);

    gemm_wmma<2, false><<<dim3(E / 128, M / 128), blk, 0, stream>>>(
        yb, atp_w + (size_t)i * E * E, atp_b + (size_t)i * E,
        x, nullptr, M, E, E, E);

    layernorm_k<<<M, blk, 0, stream>>>(x, ln2_w + (size_t)i * E, ln2_b + (size_t)i * E, hb);

    gemm_wmma<1, false><<<dim3(FF / 128, M / 128), blk, 0, stream>>>(
        hb, fc_w + (size_t)i * E * FF, fc_b + (size_t)i * FF,
        nullptr, mb, M, FF, E, FF);

    gemm_wmma<2, false><<<dim3(E / 128, M / 128), blk, 0, stream>>>(
        mb, fp_w + (size_t)i * FF * E, fp_b + (size_t)i * E,
        x, nullptr, M, E, FF, E);
  }

  layernorm_k<<<M, blk, 0, stream>>>(x, lnf_w, lnf_b, hb);

  // logits = z @ lm_w.T ; lm_w is [V,E] row-major == N-major weight.
  gemm_wmma<3, true><<<dim3((V + 127) / 128, M / 128), blk, 0, stream>>>(
      hb, lm_w, nullptr, (float*)d_out, nullptr, M, V, E, E);
}
